// DFBL_11269994185467
// MI455X (gfx1250) — matrix operand order repaired and checked by reading, not compiled, not run on hardware
//
#include <hip/hip_runtime.h>

typedef __attribute__((ext_vector_type(16))) _Float16 v16h;
typedef __attribute__((ext_vector_type(8)))  float    v8f;

#define SR_F     16000.0f
#define N_FILT   64
#define GK       401
#define KPAD     416           // taps padded to 13*32
#define KW       (KPAD/2)      // 208 u32 per filter row
#define PSTRIDE  160
#define TLEN     160000
#define BATCH    8
#define TP       1000
#define PBLK     20            // pooled outputs per workgroup
#define NPB      (TP/PBLK)     // 50
#define NT       216           // 16-wide conv time-tiles per workgroup
#define NPAIR    (NT/2)        // 108 32-wide tile pairs
#define XS_LEN   3872          // staged x length (f16)
#define FMIN_F   30.0f
#define FMAX_F   4000.0f       // SR/2*0.5

// ---------------- Kernel 1: build Gabor (f16) + Kaiser (f32) taps ----------------

__device__ inline float i0_series(float x) {
    // modified Bessel I0 via power series; converged for x <= 20 (beta clip range)
    float x2 = 0.25f * x * x;
    float term = 1.0f, sum = 1.0f;
    #pragma unroll 1
    for (int k = 1; k <= 40; ++k) {
        term *= x2 / ((float)k * (float)k);
        sum += term;
    }
    return sum;
}

__global__ __launch_bounds__(128)
void dfbl_filters(const float* __restrict__ pc, const float* __restrict__ pb,
                  const float* __restrict__ beta,
                  _Float16* __restrict__ wr, _Float16* __restrict__ wi,
                  float* __restrict__ ka) {
    const int n = blockIdx.x;

    float fc = expf(pc[n]);
    fc = fminf(fmaxf(fc, FMIN_F), FMAX_F - 10.0f);
    float xb = pb[n];
    float sp = (xb > 20.0f) ? xb : log1pf(expf(xb));
    float bw_pos = sp * 1000.0f;
    float max_bw = 2.0f * fminf(fc - FMIN_F, FMAX_F - fc);
    float bw = fminf(bw_pos, fmaxf(max_bw, 50.0f));
    float f_low  = fmaxf(fc - 0.5f * bw, FMIN_F);
    float f_high = fminf(fc + 0.5f * bw, FMAX_F);
    float sigma = 0.5f / fmaxf(f_high - f_low, 20.0f);

    float bb = fminf(fmaxf(beta[n], 1.0f), 20.0f);
    float i0b = i0_series(bb) + 1e-8f;

    for (int i = threadIdx.x; i < KPAD; i += blockDim.x) {
        _Float16 r = (_Float16)0.0f, im = (_Float16)0.0f;
        if (i < GK) {
            float t = (float)(i - 200) * (1.0f / SR_F);
            float u = t / sigma;
            float env = expf(-0.5f * u * u);
            float ph = 6.28318530717958647692f * fc * t;
            r  = (_Float16)(env * cosf(ph));
            im = (_Float16)(env * sinf(ph));
            float q = 2.0f * (float)i / 400.0f - 1.0f;
            float arg = bb * sqrtf(fmaxf(1.0f - q * q, 0.0f));
            ka[n * GK + i] = i0_series(arg) / i0b;
        }
        wr[n * KPAD + i] = r;
        wi[n * KPAD + i] = im;
    }
}

// ------- Kernel 2: fused Gabor conv (WMMA f16->f32) + |.|^2 + Kaiser pooling -------
// Grid: (50 pool-blocks, 4 filter-tiles, 8 batches); 256 threads = 8 waves.
// Each workgroup owns pooled outputs [p0, p0+20) for 16 filters of one batch.
// Each wave iteration computes a PAIR of adjacent 16-wide time tiles so the
// A-matrix (filter taps) is loaded once per 4 WMMAs.

union U16 { v16h v; unsigned int u[8]; _Float16 h[16]; };

__global__ __launch_bounds__(256)
void dfbl_conv_pool(const float* __restrict__ x,
                    const _Float16* __restrict__ wr,
                    const _Float16* __restrict__ wi,
                    const float* __restrict__ ka,
                    float* __restrict__ pooled) {
    __shared__ _Float16     xs[XS_LEN];        //  7744 B  staged signal (f16, zero padded)
    __shared__ unsigned int wr_s[16 * KW];     // 13312 B  real taps, packed pairs
    __shared__ unsigned int wi_s[16 * KW];     // 13312 B  imag taps, packed pairs
    __shared__ float        ka_s[16 * GK];     // 25664 B  kaiser taps
    __shared__ float        pool_s[16 * PBLK]; //  1280 B  pooled accumulators

    const int pblk = blockIdx.x;   // 0..49
    const int ft   = blockIdx.y;   // 0..3  (16-filter tile)
    const int b    = blockIdx.z;   // 0..7
    const int tid  = threadIdx.x;
    const int p0    = pblk * PBLK;
    const int wg_t0 = p0 * PSTRIDE - 200;  // global scal index of local t_off=0
    const int x0    = wg_t0 - 200;         // global x index of xs[0]

    // stage x (f32 -> f16, zero pad out of range)
    for (int i = tid; i < XS_LEN; i += 256) {
        int g = x0 + i;
        float v = (g >= 0 && g < TLEN) ? x[b * TLEN + g] : 0.0f;
        xs[i] = (_Float16)v;
    }
    // stage 16-filter weight tile as packed u32 (rows are KPAD-aligned => u32 aligned)
    const unsigned int* wrg = (const unsigned int*)(wr + (size_t)ft * 16 * KPAD);
    const unsigned int* wig = (const unsigned int*)(wi + (size_t)ft * 16 * KPAD);
    for (int i = tid; i < 16 * KW; i += 256) { wr_s[i] = wrg[i]; wi_s[i] = wig[i]; }
    for (int i = tid; i < 16 * GK;   i += 256) ka_s[i]  = ka[ft * 16 * GK + i];
    for (int i = tid; i < 16 * PBLK; i += 256) pool_s[i] = 0.0f;
    __syncthreads();

    const int lane  = tid & 31;
    const int wave  = tid >> 5;
    const int laneN = lane & 15;
    const int hi    = lane >> 4;            // half-wave select
    const int kbB   = hi * 16;              // B-matrix K base for this half-wave
    const int arow  = laneN * KW + 8 * hi;  // A-matrix u32 row base (contiguous K: 16*hi)

    for (int pair = wave; pair < NPAIR; pair += 8) {  // wave-uniform bound: EXEC all-1s
        const int t0 = pair * 32;
        v8f accr[2] = {{}, {}};
        v8f acci[2] = {{}, {}};

        #pragma unroll 1
        for (int ks = 0; ks < 13; ++ks) {
            U16 av, cv, bv0, bv1;
            const int kub = arow + ks * 16;
            #pragma unroll
            for (int r = 0; r < 8; ++r) {
                // K pairs: 2r (+16*hi)  => u32 idx r
                int off = r;
                av.u[r] = wr_s[kub + off];
                cv.u[r] = wi_s[kub + off];
            }
            // Toeplitz B: B[kk, n] = xs[t0 + n + ks*32 + kbB + kk] (and +16 for 2nd tile)
            const int bx = t0 + laneN + ks * 32 + kbB;
            #pragma unroll
            for (int r = 0; r < 8; ++r) {
                bv0.h[2 * r]     = xs[bx + 2 * r];
                bv0.h[2 * r + 1] = xs[bx + 2 * r + 1];
                bv1.h[2 * r]     = xs[bx + 16 + 2 * r];
                bv1.h[2 * r + 1] = xs[bx + 16 + 2 * r + 1];
            }
            accr[0] = __builtin_amdgcn_wmma_f32_16x16x32_f16(
                false, av.v, false, bv0.v, (short)0, accr[0], false, false);
            acci[0] = __builtin_amdgcn_wmma_f32_16x16x32_f16(
                false, cv.v, false, bv0.v, (short)0, acci[0], false, false);
            accr[1] = __builtin_amdgcn_wmma_f32_16x16x32_f16(
                false, av.v, false, bv1.v, (short)0, accr[1], false, false);
            acci[1] = __builtin_amdgcn_wmma_f32_16x16x32_f16(
                false, cv.v, false, bv1.v, (short)0, acci[1], false, false);
        }

        // scal = 0.5*(r^2+i^2); scatter into Kaiser pooled accumulators (LDS atomics)
        #pragma unroll
        for (int s = 0; s < 2; ++s) {
            const int toff = t0 + 16 * s + laneN;
            const int tg   = wg_t0 + toff;
            if (tg >= 0 && tg < TLEN) {      // conv output only exists for tg in [0,T)
                int plo = toff - 400;
                plo = (plo <= 0) ? 0 : ((plo + PSTRIDE - 1) / PSTRIDE);  // ceil
                int phi = toff / PSTRIDE;
                if (phi > PBLK - 1) phi = PBLK - 1;
                if (plo <= phi) {
                    #pragma unroll
                    for (int r = 0; r < 8; ++r) {
                        const int mrow = r + 8 * hi;  // C/D layout: lanes16-31 hold M=r+8
                        float rr = accr[s][r], ii = acci[s][r];
                        float scal = 0.5f * (rr * rr + ii * ii);
                        for (int p = plo; p <= phi; ++p) {
                            int j = toff - p * PSTRIDE;      // 0..400 by construction
                            atomicAdd(&pool_s[mrow * PBLK + p], scal * ka_s[mrow * GK + j]);
                        }
                    }
                }
            }
        }
    }
    __syncthreads();

    for (int i = tid; i < 16 * PBLK; i += 256) {
        int mrow = i / PBLK, p = i % PBLK;
        int n = ft * 16 + mrow;
        pooled[((size_t)b * N_FILT + n) * TP + p0 + p] = pool_s[i];
    }
}

// ---------------- Kernel 3: PCEN (sequential EMA per (b,n) row) ----------------

__global__ __launch_bounds__(256)
void dfbl_pcen(const float* __restrict__ pooled,
               const float* __restrict__ gi, const float* __restrict__ oi,
               const float* __restrict__ ei, float* __restrict__ out) {
    int id = blockIdx.x * blockDim.x + threadIdx.x;
    if (id >= BATCH * N_FILT) return;
    int n = id % N_FILT;
    float g = fminf(fmaxf(gi[n], 0.5f), 0.999f);
    float o = fminf(fmaxf(oi[n], 0.0f), 10.0f);
    float e = fminf(fmaxf(ei[n], 0.1f), 1.0f);
    float oe = powf(o, e);
    const float s = 0.025f;
    float ema = 0.0f;
    const float* src = pooled + (size_t)id * TP;
    float*       dst = out    + (size_t)id * TP;
    #pragma unroll 1
    for (int p = 0; p < TP; ++p) {
        float xv = src[p];
        ema = (1.0f - s) * ema + s * xv;
        float y = powf(xv / (ema + 1e-6f) + o, e) - oe;
        dst[p] = y * g;
    }
}

// ---------------- launch ----------------

extern "C" void kernel_launch(void* const* d_in, const int* in_sizes, int n_in,
                              void* d_out, int out_size, void* d_ws, size_t ws_size,
                              hipStream_t stream) {
    const float* x  = (const float*)d_in[0];
    const float* pc = (const float*)d_in[1];
    const float* pb = (const float*)d_in[2];
    const float* be = (const float*)d_in[3];
    const float* pg = (const float*)d_in[4];
    const float* po = (const float*)d_in[5];
    const float* pe = (const float*)d_in[6];
    float* out = (float*)d_out;

    // workspace layout (256B-aligned offsets)
    char* ws = (char*)d_ws;
    _Float16* wr     = (_Float16*)(ws + 0);        // 64*416*2   = 53248 B
    _Float16* wi     = (_Float16*)(ws + 53248);    // 53248 B
    float*    ka     = (float*)   (ws + 106496);   // 64*401*4   = 102656 B
    float*    pooled = (float*)   (ws + 209152);   // 8*64*1000*4 = 2048000 B

    dfbl_filters<<<N_FILT, 128, 0, stream>>>(pc, pb, be, wr, wi, ka);
    dfbl_conv_pool<<<dim3(NPB, 4, BATCH), 256, 0, stream>>>(x, wr, wi, ka, pooled);
    dfbl_pcen<<<2, 256, 0, stream>>>(pooled, pg, po, pe, out);
    (void)in_sizes; (void)n_in; (void)out_size; (void)ws_size;
}
